// RKHS_VNN_34557306863620
// MI455X (gfx1250) — matrix operand order repaired and checked by reading, not compiled, not run on hardware
//
#include <hip/hip_runtime.h>

// ---------------------------------------------------------------------------
// MI455X (gfx1250). Poly layers collapsed to single GEMMs (W = proj^T @ alpha,
// valid since clamp follows both einsums). Conv GEMMs on
// v_wmma_f32_16x16x32_bf16; weights staged to LDS with
// global_load_async_to_lds_b128 (ASYNCcnt). A-fragment im2col gather is fully
// branchless; f32->bf16 packing via v_perm_b32 (1 op per element pair).
//   flat = s*(CI*27) + q  ==>  t = q%27, z = s*CI + q/27, c = z/S, pos = z%S
// ---------------------------------------------------------------------------

typedef __attribute__((ext_vector_type(16))) __bf16 v16bf;
typedef __attribute__((ext_vector_type(8)))  float  v8f;
typedef unsigned __attribute__((address_space(3))) lds_u32;

__device__ __forceinline__ unsigned short f2bf(float f) {
  union { float f; unsigned u; } c; c.f = f;
  unsigned u = c.u;
  u += 0x7FFFu + ((u >> 16) & 1u);            // round-to-nearest-even
  return (unsigned short)(u >> 16);
}

// pack two floats into a bf16-pair dword: one v_add each + one v_perm_b32
__device__ __forceinline__ unsigned pack_bf16x2(float f0, float f1) {
  unsigned u0 = __float_as_uint(f0) + 0x8000u;   // round-half-up
  unsigned u1 = __float_as_uint(f1) + 0x8000u;
  return __builtin_amdgcn_perm(u1, u0, 0x07060302u);  // {u1.hi16, u0.hi16}
}

// ---------------- weight preparation ----------------------------------------
// Output layout is the WMMA B-fragment packed form, per 16-wide N tile:
//   Wp[((nt*(KPAD/2)) + kk)*16 + n] = bf16(W[2kk][nt*16+n]) | bf16(W[2kk+1][..])<<16
// so LDS staging in the GEMM is a straight contiguous copy.

__global__ void fuse_poly_w(const float* __restrict__ proj,
                            const float* __restrict__ alpha,
                            unsigned int* __restrict__ Wp,
                            int K, int CO, int KPAD, int NPAD) {
  int ntiles = NPAD / 16;
  int total  = ntiles * (KPAD / 2) * 16;
  int idx = blockIdx.x * blockDim.x + threadIdx.x;
  if (idx >= total) return;
  int nt = idx / ((KPAD / 2) * 16);
  int r  = idx - nt * (KPAD / 2) * 16;
  int kk = r >> 4, n = r & 15;
  int gn = nt * 16 + n;
  float v0 = 0.f, v1 = 0.f;
  if (gn < CO) {
    int k0 = 2 * kk, k1 = 2 * kk + 1;
    if (k0 < K) for (int p = 0; p < 64; ++p) v0 += proj[p * K + k0] * alpha[p * CO + gn];
    if (k1 < K) for (int p = 0; p < 64; ++p) v1 += proj[p * K + k1] * alpha[p * CO + gn];
  }
  Wp[idx] = (unsigned)f2bf(v0) | ((unsigned)f2bf(v1) << 16);
}

__global__ void cast_pw_w(const float* __restrict__ w,  // (CO x CI) row-major
                          unsigned int* __restrict__ Wp,
                          int CI, int CO, int KPAD, int NPAD) {
  int ntiles = NPAD / 16;
  int total  = ntiles * (KPAD / 2) * 16;
  int idx = blockIdx.x * blockDim.x + threadIdx.x;
  if (idx >= total) return;
  int nt = idx / ((KPAD / 2) * 16);
  int r  = idx - nt * (KPAD / 2) * 16;
  int kk = r >> 4, n = r & 15;
  int gn = nt * 16 + n;
  int k0 = 2 * kk, k1 = 2 * kk + 1;
  float v0 = (gn < CO && k0 < CI) ? w[gn * CI + k0] : 0.f;
  float v1 = (gn < CO && k1 < CI) ? w[gn * CI + k1] : 0.f;
  Wp[idx] = (unsigned)f2bf(v0) | ((unsigned)f2bf(v1) << 16);
}

// ---------------- main WMMA conv GEMM ---------------------------------------
// Y[b, n, s] (+)= [clamp(1+.,0)] ( sum_k A[m,k] * W[k,n] )  (+ bias[n])
// Each wave: one 16-row M tile x NT 16-col N tiles (A fetched once, NT WMMAs).
template <int CI, int DD, int HH, int WW, int CO, int NT,
          bool PATCH, bool CLAMP, bool ACC, bool HASB>
__global__ __launch_bounds__(256)
void conv_gemm_wmma(const float* __restrict__ X,
                    const unsigned int* __restrict__ Wp,
                    const float* __restrict__ bias,
                    float* __restrict__ Y) {
  constexpr int S    = DD * HH * WW;
  constexpr int M    = 4 * S;
  constexpr int K    = PATCH ? CI * 27 : CI;
  constexpr int KPAD = ((K + 31) / 32) * 32;
  constexpr int WPT  = (KPAD / 2) * 16;      // packed words per N tile
  constexpr int WT   = NT * WPT;             // words staged per block

  __shared__ unsigned int ldsB[WT];

  const int tid    = threadIdx.x;
  const int ntile0 = blockIdx.y * NT;

  // ---- async copy of the packed weight slice into LDS (ASYNCcnt path) ----
  {
    unsigned ldsBase = (unsigned)(size_t)(lds_u32*)ldsB;
    const unsigned int* src = Wp + (size_t)ntile0 * WPT;
    for (int i = tid; i < WT / 4; i += 256) {
      unsigned dst = ldsBase + (unsigned)i * 16u;
      const unsigned int* sp = src + i * 4;
      asm volatile("global_load_async_to_lds_b128 %0, %1, off"
                   :: "v"(dst), "v"(sp) : "memory");
    }
    asm volatile("s_wait_asynccnt 0x0" ::: "memory");
  }
  __syncthreads();

  const int wave = tid >> 5;
  const int lane = tid & 31;
  const int hi16 = (lane & 16) ? 1 : 0;
  const int m0   = blockIdx.x * 128 + wave * 16;
  const int mrow = m0 + (lane & 15);
  const bool mvalid = (mrow < M);
  const int b = mrow / S;
  const int s = mrow - b * S;

  // per-lane constants for the de-scrambled im2col decomposition
  int cb = 0, pb = 0, d0 = 0, h0 = 0, w0 = 0;
  if (PATCH) {
    int zb = s * CI;
    cb = zb / S;  pb = zb - cb * S;
    d0 = pb / (HH * WW);
    int rr = pb - d0 * HH * WW;
    h0 = rr / WW; w0 = rr - h0 * WW;
  }

  // fully branchless A-element fetch (selects only, no EXEC divergence)
  auto fetchA = [&](int q, bool ok) -> float {
    if (!PATCH) {
      size_t aidx = ok ? ((size_t)(b * CI + q) * S + s) : 0;
      float v = X[aidx];
      return ok ? v : 0.f;
    }
    int qc = q / 27;              // const divisions of small values
    int t  = q - 27 * qc;
    int di = t / 9;  int tj = t - 9 * di;
    int dj = tj / 3; int dk = tj - 3 * dj;
    int pos   = pb + qc;
    int carry = (pos >= S) ? 1 : 0;   // channel-boundary carry (pos-S < CI)
    int c  = cb + carry;
    int bd = carry ? 0 : d0;
    int bh = carry ? 0 : h0;
    int bw = carry ? 0 : w0;
    int qq = carry ? (pos - S) : qc;
    int v2 = bw + qq;
    int wq = v2 / WW; int ww = v2 - wq * WW;
    int hh = bh + wq;
    int ca = (hh >= HH) ? 1 : 0;
    hh -= ca * HH;
    int dd = bd + ca;
    int zd = dd + di - 1, zh = hh + dj - 1, zw = ww + dk - 1;
    bool inb = ok & ((unsigned)zd < (unsigned)DD) &
                    ((unsigned)zh < (unsigned)HH) &
                    ((unsigned)zw < (unsigned)WW);
    size_t aidx = inb ? ((size_t)(b * CI + c) * S + (zd * HH + zh) * WW + zw)
                      : 0;
    float v = X[aidx];            // index 0 is always a safe load
    return inb ? v : 0.f;
  };

  union BFA { v16bf v; unsigned int u32[8]; };
  union BFB { v16bf v; unsigned int u32[8]; };

  v8f acc[NT];
#pragma unroll
  for (int nt = 0; nt < NT; ++nt) acc[nt] = (v8f){};

  for (int k0 = 0; k0 < KPAD; k0 += 32) {
    BFA A;
#pragma unroll
    for (int p2 = 0; p2 < 8; ++p2) {
      // A k order used here (B matches): lanes 0-15 K in [0..15], lanes 16-31 K in [16..31].
      // element pair (2*p2, 2*p2+1) lives in the same K-half
      int e0   = 2 * p2;
      int kOff = e0 + hi16 * 16;
      int q0   = k0 + kOff;
      bool ok0 = mvalid & (q0 < K);
      bool ok1 = mvalid & ((q0 + 1) < K);
      float f0 = fetchA(q0, ok0);
      float f1 = fetchA(q0 + 1, ok1);
      A.u32[p2] = pack_bf16x2(f0, f1);   // v_perm_b32 packing
    }
    int kkBase = (k0 >> 1) + hi16 * 8;
#pragma unroll
    for (int nt = 0; nt < NT; ++nt) {
      BFB Bf;
#pragma unroll
      for (int i = 0; i < 8; ++i)
        Bf.u32[i] = ldsB[nt * WPT + (kkBase + i) * 16 + (lane & 15)];
      acc[nt] = __builtin_amdgcn_wmma_f32_16x16x32_bf16(
          false, A.v, false, Bf.v, (short)0, acc[nt], false, false);
    }
  }

#pragma unroll
  for (int nt = 0; nt < NT; ++nt) {
    const int n = (ntile0 + nt) * 16 + (lane & 15);
    if (n < CO) {
      float bv = HASB ? bias[n] : 0.f;
#pragma unroll
      for (int r = 0; r < 8; ++r) {
        int mr = m0 + r + hi16 * 8;   // C/D layout: lanes 16-31 hold rows M+8
        if (mr < M) {
          int bb = mr / S;
          int s2 = mr - bb * S;
          float v = acc[nt][r];
          if (CLAMP) v = fmaxf(1.f + v, 0.f);
          v += bv;
          size_t oi = (size_t)(bb * CO + n) * S + s2;
          if (ACC) Y[oi] += v; else Y[oi] = v;
        }
      }
    }
  }
}

// ---------------- depthwise 3x3x3 (branchless taps) -------------------------
template <int C, int DD, int HH, int WW>
__global__ void dwconv3(const float* __restrict__ X, const float* __restrict__ Wd,
                        const float* __restrict__ bias, float* __restrict__ Y) {
  constexpr int S = DD * HH * WW;
  int idx = blockIdx.x * blockDim.x + threadIdx.x;
  if (idx >= 4 * C * S) return;
  int bb = idx / (C * S); int r = idx - bb * C * S;
  int c  = r / S;         int sp = r - c * S;
  int d  = sp / (HH * WW); int r2 = sp - d * HH * WW;
  int h  = r2 / WW;        int w  = r2 - h * WW;
  const float* xb = X + (size_t)(bb * C + c) * S;
  float sum = bias[c];
#pragma unroll
  for (int t = 0; t < 27; ++t) {
    int di = t / 9; int tj = t - di * 9;
    int dj = tj / 3; int dk = tj - dj * 3;
    int zd = d + di - 1, zh = h + dj - 1, zw = w + dk - 1;
    bool inb = ((unsigned)zd < (unsigned)DD) & ((unsigned)zh < (unsigned)HH) &
               ((unsigned)zw < (unsigned)WW);
    size_t ai = inb ? (size_t)((zd * HH + zh) * WW + zw) : 0;
    float xv = xb[ai];
    sum += inb ? xv * Wd[c * 27 + t] : 0.f;
  }
  Y[idx] = sum;
}

// ---------------- maxpool(1,2,2) + BN batch stats ---------------------------
template <int C, int DD, int HH, int WW>
__global__ __launch_bounds__(256)
void pool_bn_stats(const float* __restrict__ X, float* __restrict__ Y,
                   float* __restrict__ sums /* [2*C] */) {
  constexpr int HO = HH / 2, WO = WW / 2;
  constexpr int SO = DD * HO * WO;
  __shared__ float ls[2 * C];
  for (int i = threadIdx.x; i < 2 * C; i += 256) ls[i] = 0.f;
  __syncthreads();

  int idx = blockIdx.x * blockDim.x + threadIdx.x;
  if (idx < 4 * C * SO) {
    int bb = idx / (C * SO); int r = idx - bb * C * SO;
    int c  = r / SO;         int sp = r - c * SO;
    int d  = sp / (HO * WO); int r2 = sp - d * HO * WO;
    int ho = r2 / WO;        int wo = r2 - ho * WO;
    const float* xb = X + ((size_t)(bb * C + c) * DD + d) * HH * WW;
    int h = 2 * ho, w = 2 * wo;
    float m = fmaxf(fmaxf(xb[h * WW + w], xb[h * WW + w + 1]),
                    fmaxf(xb[(h + 1) * WW + w], xb[(h + 1) * WW + w + 1]));
    Y[idx] = m;
    atomicAdd(&ls[c], m);          // LDS ds_add_f32
    atomicAdd(&ls[C + c], m * m);
  }
  __syncthreads();
  for (int i = threadIdx.x; i < 2 * C; i += 256)
    if (ls[i] != 0.f) atomicAdd(&sums[i], ls[i]);   // 1 global atomic/chan/blk
}

__global__ void bn_finalize(const float* __restrict__ sums,
                            const float* __restrict__ gamma,
                            const float* __restrict__ beta,
                            float* __restrict__ ss, int C, float invCnt) {
  int c = threadIdx.x;
  if (c >= C) return;
  float mean = sums[c] * invCnt;
  float var  = sums[C + c] * invCnt - mean * mean;
  float sc   = gamma[c] * rsqrtf(var + 1e-5f);
  ss[c]     = sc;
  ss[C + c] = beta[c] - mean * sc;
}

template <int C, int S>
__global__ void bn_apply(float* __restrict__ Y, const float* __restrict__ ss) {
  int idx = blockIdx.x * blockDim.x + threadIdx.x;
  if (idx >= 4 * C * S) return;
  int c = (idx / S) % C;
  Y[idx] = Y[idx] * ss[c] + ss[C + c];
}

__global__ void zero_f32(float* __restrict__ p, int n) {
  int i = blockIdx.x * blockDim.x + threadIdx.x;
  if (i < n) p[i] = 0.f;
}

// ---------------- fully connected (M=4, reduction kernels) ------------------
__global__ void fc_kernel(const float* __restrict__ Xf, const float* __restrict__ W,
                          const float* __restrict__ b, float* __restrict__ Y,
                          int K, int N, int doRelu) {
  __shared__ float red[128];
  int n = blockIdx.x, bb = blockIdx.y;
  float a = 0.f;
  for (int k = threadIdx.x; k < K; k += 128) a += Xf[bb * K + k] * W[n * K + k];
  red[threadIdx.x] = a;
  __syncthreads();
  for (int s = 64; s > 0; s >>= 1) {
    if (threadIdx.x < s) red[threadIdx.x] += red[threadIdx.x + s];
    __syncthreads();
  }
  if (threadIdx.x == 0) {
    float v = red[0] + b[n];
    if (doRelu) v = fmaxf(v, 0.f);
    Y[bb * N + n] = v;
  }
}

// ---------------------------------------------------------------------------

extern "C" void kernel_launch(void* const* d_in, const int* in_sizes, int n_in,
                              void* d_out, int out_size, void* d_ws, size_t ws_size,
                              hipStream_t stream) {
  (void)in_sizes; (void)n_in; (void)out_size; (void)ws_size;
  const float* x      = (const float*)d_in[0];
  const float* proj1  = (const float*)d_in[1];
  const float* proj2  = (const float*)d_in[2];
  const float* proj3  = (const float*)d_in[3];
  const float* alpha1 = (const float*)d_in[4];
  const float* alpha2 = (const float*)d_in[5];
  const float* alpha3 = (const float*)d_in[6];
  const float* dw4    = (const float*)d_in[7];
  const float* dwb4   = (const float*)d_in[8];
  const float* pw4    = (const float*)d_in[9];
  const float* pwb4   = (const float*)d_in[10];
  const float* dw5    = (const float*)d_in[11];
  const float* dwb5   = (const float*)d_in[12];
  const float* pw5    = (const float*)d_in[13];
  const float* pwb5   = (const float*)d_in[14];
  const float* res2   = (const float*)d_in[15];
  const float* res3   = (const float*)d_in[16];
  const float* res4   = (const float*)d_in[17];
  const float* res5   = (const float*)d_in[18];
  const float* w_fc1  = (const float*)d_in[19];
  const float* b_fc1  = (const float*)d_in[20];
  const float* w_fc2  = (const float*)d_in[21];
  const float* b_fc2  = (const float*)d_in[22];
  const float* gamma[5] = {(const float*)d_in[23], (const float*)d_in[25],
                           (const float*)d_in[27], (const float*)d_in[29],
                           (const float*)d_in[31]};
  const float* beta[5]  = {(const float*)d_in[24], (const float*)d_in[26],
                           (const float*)d_in[28], (const float*)d_in[30],
                           (const float*)d_in[32]};

  // -------- workspace carve-out --------
  char* ws = (char*)d_ws;
  size_t off = 0;
  auto alloc = [&](size_t bytes) {
    void* p = ws + off;
    off = (off + bytes + 255) & ~(size_t)255;
    return p;
  };
  unsigned int* w1    = (unsigned int*)alloc(96  * 16  * 2);
  unsigned int* w2    = (unsigned int*)alloc(448 * 32  * 2);
  unsigned int* w3    = (unsigned int*)alloc(672 * 48  * 2);
  unsigned int* wres2 = (unsigned int*)alloc(32  * 32  * 2);
  unsigned int* wres3 = (unsigned int*)alloc(32  * 48  * 2);
  unsigned int* wres4 = (unsigned int*)alloc(64  * 64  * 2);
  unsigned int* wres5 = (unsigned int*)alloc(64  * 128 * 2);
  unsigned int* wpw4  = (unsigned int*)alloc(64  * 64  * 2);
  unsigned int* wpw5  = (unsigned int*)alloc(64  * 128 * 2);
  float* stats = (float*)alloc(256 * 4);
  float* ss    = (float*)alloc(256 * 4);
  float* act0  = (float*)alloc((size_t)4 * 16 * 16 * 112 * 112 * 4); // pre-pool
  float* act1  = (float*)alloc((size_t)4 * 16 * 16 * 56  * 56  * 4); // pooled
  float* act2  = (float*)alloc((size_t)4 * 48 * 16 * 14  * 14  * 4); // dw tmp
  float* fcbuf = (float*)alloc(4 * 256 * 4);

  auto cdiv = [](int a, int b) { return (a + b - 1) / b; };

  // -------- weight prep (packed B-fragment layout) --------
  fuse_poly_w<<<cdiv(1 * 48 * 16, 256), 256, 0, stream>>>(proj1, alpha1, w1, 81, 16, 96, 16);
  fuse_poly_w<<<cdiv(2 * 224 * 16, 256), 256, 0, stream>>>(proj2, alpha2, w2, 432, 24, 448, 32);
  fuse_poly_w<<<cdiv(3 * 336 * 16, 256), 256, 0, stream>>>(proj3, alpha3, w3, 648, 48, 672, 48);
  cast_pw_w<<<cdiv(2 * 16 * 16, 256), 256, 0, stream>>>(res2, wres2, 16, 24, 32, 32);
  cast_pw_w<<<cdiv(3 * 16 * 16, 256), 256, 0, stream>>>(res3, wres3, 24, 48, 32, 48);
  cast_pw_w<<<cdiv(4 * 32 * 16, 256), 256, 0, stream>>>(res4, wres4, 48, 64, 64, 64);
  cast_pw_w<<<cdiv(8 * 32 * 16, 256), 256, 0, stream>>>(res5, wres5, 64, 128, 64, 128);
  cast_pw_w<<<cdiv(4 * 32 * 16, 256), 256, 0, stream>>>(pw4, wpw4, 48, 64, 64, 64);
  cast_pw_w<<<cdiv(8 * 32 * 16, 256), 256, 0, stream>>>(pw5, wpw5, 64, 128, 64, 128);

  // -------- layer 1: poly conv (3->16) @ 16x112x112 --------
  conv_gemm_wmma<3, 16, 112, 112, 16, 1, true, true, false, false>
      <<<dim3(802816 / 128, 1), 256, 0, stream>>>(x, w1, nullptr, act0);
  zero_f32<<<1, 256, 0, stream>>>(stats, 256);
  pool_bn_stats<16, 16, 112, 112>
      <<<cdiv(4 * 16 * 16 * 56 * 56, 256), 256, 0, stream>>>(act0, act1, stats);
  bn_finalize<<<1, 128, 0, stream>>>(stats, gamma[0], beta[0], ss, 16,
                                     1.f / (4.f * 16 * 56 * 56));
  bn_apply<16, 16 * 56 * 56>
      <<<cdiv(4 * 16 * 16 * 56 * 56, 256), 256, 0, stream>>>(act1, ss);

  // -------- layer 2: residual pw (16->24) + poly conv @ 16x56x56 --------
  conv_gemm_wmma<16, 16, 56, 56, 24, 2, false, false, false, false>
      <<<dim3(200704 / 128, 1), 256, 0, stream>>>(act1, wres2, nullptr, act0);
  conv_gemm_wmma<16, 16, 56, 56, 24, 2, true, true, true, false>
      <<<dim3(200704 / 128, 1), 256, 0, stream>>>(act1, w2, nullptr, act0);
  zero_f32<<<1, 256, 0, stream>>>(stats, 256);
  pool_bn_stats<24, 16, 56, 56>
      <<<cdiv(4 * 24 * 16 * 28 * 28, 256), 256, 0, stream>>>(act0, act1, stats);
  bn_finalize<<<1, 128, 0, stream>>>(stats, gamma[1], beta[1], ss, 24,
                                     1.f / (4.f * 16 * 28 * 28));
  bn_apply<24, 16 * 28 * 28>
      <<<cdiv(4 * 24 * 16 * 28 * 28, 256), 256, 0, stream>>>(act1, ss);

  // -------- layer 3: residual pw (24->48) + poly conv @ 16x28x28 --------
  conv_gemm_wmma<24, 16, 28, 28, 48, 3, false, false, false, false>
      <<<dim3(50176 / 128, 1), 256, 0, stream>>>(act1, wres3, nullptr, act0);
  conv_gemm_wmma<24, 16, 28, 28, 48, 3, true, true, true, false>
      <<<dim3(50176 / 128, 1), 256, 0, stream>>>(act1, w3, nullptr, act0);
  zero_f32<<<1, 256, 0, stream>>>(stats, 256);
  pool_bn_stats<48, 16, 28, 28>
      <<<cdiv(4 * 48 * 16 * 14 * 14, 256), 256, 0, stream>>>(act0, act1, stats);
  bn_finalize<<<1, 128, 0, stream>>>(stats, gamma[2], beta[2], ss, 48,
                                     1.f / (4.f * 16 * 14 * 14));
  bn_apply<48, 16 * 14 * 14>
      <<<cdiv(4 * 48 * 16 * 14 * 14, 256), 256, 0, stream>>>(act1, ss);

  // -------- layer 4: res pw (48->64) + dw3 + pw (48->64) @ 16x14x14 --------
  conv_gemm_wmma<48, 16, 14, 14, 64, 4, false, false, false, false>
      <<<dim3(cdiv(12544, 128), 1), 256, 0, stream>>>(act1, wres4, nullptr, act0);
  dwconv3<48, 16, 14, 14>
      <<<cdiv(4 * 48 * 16 * 14 * 14, 256), 256, 0, stream>>>(act1, dw4, dwb4, act2);
  conv_gemm_wmma<48, 16, 14, 14, 64, 4, false, false, true, true>
      <<<dim3(cdiv(12544, 128), 1), 256, 0, stream>>>(act2, wpw4, pwb4, act0);
  zero_f32<<<1, 256, 0, stream>>>(stats, 256);
  pool_bn_stats<64, 16, 14, 14>
      <<<cdiv(4 * 64 * 16 * 7 * 7, 256), 256, 0, stream>>>(act0, act1, stats);
  bn_finalize<<<1, 128, 0, stream>>>(stats, gamma[3], beta[3], ss, 64,
                                     1.f / (4.f * 16 * 7 * 7));
  bn_apply<64, 16 * 7 * 7>
      <<<cdiv(4 * 64 * 16 * 7 * 7, 256), 256, 0, stream>>>(act1, ss);

  // -------- layer 5: res pw (64->128) + dw3 + pw (64->128) @ 16x7x7 --------
  conv_gemm_wmma<64, 16, 7, 7, 128, 4, false, false, false, false>
      <<<dim3(cdiv(3136, 128), 2), 256, 0, stream>>>(act1, wres5, nullptr, act0);
  dwconv3<64, 16, 7, 7>
      <<<cdiv(4 * 64 * 16 * 7 * 7, 256), 256, 0, stream>>>(act1, dw5, dwb5, act2);
  conv_gemm_wmma<64, 16, 7, 7, 128, 4, false, false, true, true>
      <<<dim3(cdiv(3136, 128), 2), 256, 0, stream>>>(act2, wpw5, pwb5, act0);
  zero_f32<<<1, 256, 0, stream>>>(stats, 256);
  pool_bn_stats<128, 16, 7, 7>
      <<<cdiv(4 * 128 * 16 * 3 * 3, 256), 256, 0, stream>>>(act0, act1, stats);
  bn_finalize<<<1, 128, 0, stream>>>(stats, gamma[4], beta[4], ss, 128,
                                     1.f / (4.f * 16 * 3 * 3));
  bn_apply<128, 16 * 3 * 3>
      <<<cdiv(4 * 128 * 16 * 3 * 3, 256), 256, 0, stream>>>(act1, ss);

  // -------- classifier --------
  fc_kernel<<<dim3(256, 4), 128, 0, stream>>>(act1, w_fc1, b_fc1, fcbuf,
                                              18432, 256, 1);
  fc_kernel<<<dim3(101, 4), 128, 0, stream>>>(fcbuf, w_fc2, b_fc2,
                                              (float*)d_out, 256, 101, 0);
}
